// ProposalRefineModule_21835613733329
// MI455X (gfx1250) — matrix operand order repaired and checked by reading, not compiled, not run on hardware
//
#include <hip/hip_runtime.h>
#include <hip/hip_bf16.h>

// ---------------------------------------------------------------------------
// Types for CDNA5 WMMA (wave32, 16x16x32 bf16)
// ---------------------------------------------------------------------------
typedef __attribute__((ext_vector_type(16))) __bf16 v16bf;
typedef __attribute__((ext_vector_type(8)))  float  v8f;

#define DEV __device__ __forceinline__

DEV v8f wmma_bf16(v16bf a, v16bf b, v8f c) {
  return __builtin_amdgcn_wmma_f32_16x16x32_bf16(false, a, false, b, (short)0, c,
                                                 false, false);
}

// surface offsets (6x3) followed by line offsets (12x3)
__constant__ float OFFTAB[54] = {
    0,0,0.5f,  0,0,-0.5f,  0,0.5f,0,  0,-0.5f,0,  0.5f,0,0,  -0.5f,0,0,
    0.5f,0.5f,0,  -0.5f,0.5f,0,  0.5f,-0.5f,0,  -0.5f,-0.5f,0,
    0.5f,0,0.5f,  -0.5f,0,0.5f,  0.5f,0,-0.5f,  -0.5f,0,-0.5f,
    0,0.5f,0.5f,  0,0.5f,-0.5f,  0,-0.5f,0.5f,  0,-0.5f,-0.5f};

// ---------------------------------------------------------------------------
// Flat-param offsets (pytree order: dict keys sorted; cbr leaves W,b,beta,g)
// ---------------------------------------------------------------------------
// line_agg[0] W@0 b@1024 beta@1056 g@1088 ; line_agg[1] W@1120 b@2144 beta@2176 g@2208
// matching_conv W@2240 b@3264 beta@3296 g@3328 ; matching_pred W@3360 b@3424
// matching_sem_conv W@3426 b@4450 beta@4482 g@4514 ; matching_sem_pred W@4546 b@4610
// proposal_convs[0] W@4612 b@78340 beta@78468 g@78596
// proposal_convs[1] W@78724 b@95108 beta@95236 g@95364
// proposal_convs[2] W@95492 b@111876 beta@112004 g@112132
// proposal_final W@112260 b@124676
// sa_line[0] W@124773 b@143077 beta@143205 g@143333
// sa_line[1] W@143461 b@151653 beta@151717 g@151781
// sa_line[2] W@151845 b@153893 beta@153925 g@153957
// sa_surf[0] W@153989 b@171525 beta@171653 g@171781
// sa_surf[1] W@171909 b@180101 beta@180165 g@180229
// sa_surf[2] W@180293 b@182341 beta@182373 g@182405
// surf_agg[0] W@182437 b@183461 beta@183493 g@183525
// surf_agg[1] W@183557 b@184581 beta@184613 g@184645

// d_out regions (floats): pred@0 (4*97*256), match@99328, match_sem@136192, sem@173056
#define OUT_MS   99328
#define OUT_MSS  136192
#define OUT_SEM  173056

// workspace layout: bf16 weight fragments first (half offsets), then f32/bf16 buffers
#define WH_SURF_L1 0
#define WH_LINE_L1 20480
#define WH_SURF_L2 40960
#define WH_LINE_L2 49152
#define WH_SURF_L3 57344
#define WH_LINE_L3 59392
#define WH_P1      61440
#define WH_P2      135168
#define WH_P3      151552
#define WH_PF      167936
#define WSB_SF     364544     // float, 4*32*1536
#define WSB_LF     1150976    // float, 4*32*3072
#define WSB_CF     2723840    // bf16,  4*256*576

// ---------------------------------------------------------------------------
// Prep: fold BN scale into weights, convert to bf16, swizzle into WMMA
// A fragments: row = lane % 16, contiguous k = 16*(lane/16) + e (same order as B reads).
// ---------------------------------------------------------------------------
__global__ void prep_wfrag(const float* __restrict__ W, const float* __restrict__ g,
                           int cout, int cin, int MT, int KC,
                           __bf16* __restrict__ dst) {
  int idx = blockIdx.x * 256 + threadIdx.x;
  int tot = MT * KC * 512;
  if (idx >= tot) return;
  int frag = idx >> 9, within = idx & 511;
  int lane = within >> 4, e = within & 15;
  int mt = frag / KC, kc = frag % KC;
  int row = mt * 16 + (lane & 15);
  int h = lane >> 4;
  int pair = e >> 1, lo = e & 1;
  int grp = pair >> 2, w4 = pair & 3;
  int k = kc * 32 + h * 16 + grp * 8 + w4 * 2 + lo;
  float v = 0.f;
  if (row < cout && k < cin) {
    v = W[row * cin + k];
    if (g) v *= g[row] * rsqrtf(1.f + 1e-5f);
  }
  dst[idx] = (__bf16)v;
}

// ---------------------------------------------------------------------------
// SA kernel: ball query + gather + 3-layer WMMA MLP + max-pool.
// One workgroup (8 waves) handles one batch and 4 query points (128 columns).
// ---------------------------------------------------------------------------
template <int IS_LINE>
__global__ __launch_bounds__(256) void sa_kernel(
    const float* __restrict__ prm,
    const float* __restrict__ xyzA, const float* __restrict__ xyzB,
    const float* __restrict__ featA, const float* __restrict__ featB,
    const float* __restrict__ proposal,
    const __bf16* __restrict__ wf,
    float* __restrict__ sfout) {
  constexpr int N    = IS_LINE ? 1024 : 2048;
  constexpr int Q    = IS_LINE ? 3072 : 1536;
  constexpr int FPP  = IS_LINE ? 12 : 6;
  constexpr int FOFF = IS_LINE ? 15 : 9;
  constexpr int L1F  = IS_LINE ? WH_LINE_L1 : WH_SURF_L1;
  constexpr int L2F  = IS_LINE ? WH_LINE_L2 : WH_SURF_L2;
  constexpr int L3F  = IS_LINE ? WH_LINE_L3 : WH_SURF_L3;
  constexpr int B1 = IS_LINE ? 143077 : 171525, T1 = IS_LINE ? 143205 : 171653,
                G1 = IS_LINE ? 143333 : 171781;
  constexpr int B2 = IS_LINE ? 151653 : 180101, T2 = IS_LINE ? 151717 : 180165,
                G2 = IS_LINE ? 151781 : 180229;
  constexpr int B3 = IS_LINE ? 153893 : 182341, T3 = IS_LINE ? 153925 : 182373,
                G3 = IS_LINE ? 153957 : 182405;

  __shared__ __align__(16) unsigned char smem[49152];
  __shared__ int sidx[4][32];
  __shared__ float sqc[4][3];
  __bf16* Abuf = (__bf16*)smem;                // [128 cols][160 K]
  __bf16* Bbuf = (__bf16*)smem;                // [128 cols][128 K] (aliases Abuf)
  __bf16* Cbuf = (__bf16*)(smem + 32768);      // [128 cols][64 K]
  float*  Dbuf = (float*)smem;                 // [32 ch][128 cols] (aliases Bbuf)

  const int tid = threadIdx.x, w = tid >> 5, lane = tid & 31;
  const int b = blockIdx.x / (Q / 4);
  const int qbase = (blockIdx.x % (Q / 4)) * 4;
  const float RS = rsqrtf(1.f + 1e-5f);

  // ---- Phase 1: ball query (waves 0..3, one query each) ----
  if (w < 4) {
    int qi = qbase + w;
    int p = qi / FPP, f = qi % FPP;
    const float* pr = proposal + (size_t)(b * 256 + p) * 7;
    const float* ofs = OFFTAB + (IS_LINE ? 18 : 0) + f * 3;
    float qc0 = pr[0] + ofs[0] * pr[3];
    float qc1 = pr[1] + ofs[1] * pr[4];
    float qc2 = pr[2] + ofs[2] * pr[5];
    if (lane == 0) { sqc[w][0] = qc0; sqc[w][1] = qc1; sqc[w][2] = qc2; }
    int cnt = 0;
    for (int base = 0; base < N && cnt < 32; base += 32) {
      int j = base + lane;
      const float* pt = (IS_LINE || j < 1024)
                            ? xyzA + (size_t)(b * 1024 + (j & 1023)) * 3
                            : xyzB + (size_t)(b * 1024 + (j - 1024)) * 3;
      float dx = pt[0] - qc0, dy = pt[1] - qc1, dz = pt[2] - qc2;
      bool inside = (dx * dx + dy * dy + dz * dz) < 0.25f;
      unsigned mask = __builtin_amdgcn_ballot_w32(inside);
      int pos = cnt + __popc(mask & ((1u << lane) - 1u));
      if (inside && pos < 32) sidx[w][pos] = j;
      cnt += __popc(mask);
    }
    if (cnt > 32) cnt = 32;
    int pad = (cnt > 0) ? sidx[w][0] : 0;
    if (lane >= cnt) sidx[w][lane] = pad;
  }
  __syncthreads();

  // ---- Phase 2: gather into LDS, [col][K]-contiguous (B-fragment friendly) ----
  {
    int col = tid & 127;
    int chStart = (tid >> 7) * 80;
    int q = col >> 5;
    int j = sidx[q][col & 31];
    const float* feat;
    int fj;
    if (!IS_LINE && j >= 1024) { feat = featB; fj = j - 1024; }
    else                       { feat = featA; fj = j & 1023; }
    const float* pt = (IS_LINE || j < 1024)
                          ? xyzA + (size_t)(b * 1024 + (j & 1023)) * 3
                          : xyzB + (size_t)(b * 1024 + (j - 1024)) * 3;
    float px = pt[0], py = pt[1], pz = pt[2];
    float qx = sqc[q][0], qy = sqc[q][1], qz = sqc[q][2];
    __bf16* dst = Abuf + col * 160;
    for (int ch = chStart; ch < chStart + 80; ++ch) {
      float v;
      if (ch == 0)      v = (px - qx) * 2.f;
      else if (ch == 1) v = (py - qy) * 2.f;
      else if (ch == 2) v = (pz - qz) * 2.f;
      else if (ch >= FOFF && ch < FOFF + 128)
        v = feat[(size_t)(b * 128 + (ch - FOFF)) * 1024 + fj];
      else v = 0.f;
      dst[ch] = (__bf16)v;
    }
  }
  __syncthreads();

  const int cb = lane & 15, kh = (lane >> 4) * 16, hi = (lane >> 4) * 8;

  // ---- Layer 1: 128 x (160) x 128 ----
  v8f acc[8] = {};
  {
    int mt = w;
#pragma unroll
    for (int kc = 0; kc < 5; ++kc) {
      v16bf a = *(const v16bf*)(wf + L1F + ((mt * 5 + kc) * 32 + lane) * 16);
      int kb = kc * 32 + kh;
#pragma unroll
      for (int nt = 0; nt < 8; ++nt) {
        v16bf bb = *(const v16bf*)(Abuf + (nt * 16 + cb) * 160 + kb);
        acc[nt] = wmma_bf16(a, bb, acc[nt]);
      }
    }
  }
  __syncthreads();  // Abuf dead; safe to write aliased Bbuf
  {
    int mt = w;
    float fb[8];
#pragma unroll
    for (int r = 0; r < 8; ++r) {
      int row = mt * 16 + r + hi;
      fb[r] = prm[G1 + row] * RS * prm[B1 + row] + prm[T1 + row];
    }
#pragma unroll
    for (int nt = 0; nt < 8; ++nt)
#pragma unroll
      for (int r = 0; r < 8; ++r) {
        int row = mt * 16 + r + hi;
        float v = fmaxf(acc[nt][r] + fb[r], 0.f);
        Bbuf[(nt * 16 + cb) * 128 + row] = (__bf16)v;
      }
  }
  __syncthreads();

  // ---- Layer 2: 64 x 128 x 128 ----
  v8f acc2[4] = {};
  {
    int mt = w & 3, ntb = (w >> 2) * 4;
#pragma unroll
    for (int kc = 0; kc < 4; ++kc) {
      v16bf a = *(const v16bf*)(wf + L2F + ((mt * 4 + kc) * 32 + lane) * 16);
      int kb = kc * 32 + kh;
#pragma unroll
      for (int i = 0; i < 4; ++i) {
        v16bf bb = *(const v16bf*)(Bbuf + ((ntb + i) * 16 + cb) * 128 + kb);
        acc2[i] = wmma_bf16(a, bb, acc2[i]);
      }
    }
    float fb[8];
#pragma unroll
    for (int r = 0; r < 8; ++r) {
      int row = mt * 16 + r + hi;
      fb[r] = prm[G2 + row] * RS * prm[B2 + row] + prm[T2 + row];
    }
#pragma unroll
    for (int i = 0; i < 4; ++i)
#pragma unroll
      for (int r = 0; r < 8; ++r) {
        int row = mt * 16 + r + hi;
        float v = fmaxf(acc2[i][r] + fb[r], 0.f);
        Cbuf[((ntb + i) * 16 + cb) * 64 + row] = (__bf16)v;
      }
  }
  __syncthreads();

  // ---- Layer 3: 32 x 64 x 128, then max-pool over 32 samples ----
  v8f acc3[2] = {};
  {
#pragma unroll
    for (int kc = 0; kc < 2; ++kc) {
      int kb = kc * 32 + kh;
      v16bf bb = *(const v16bf*)(Cbuf + (w * 16 + cb) * 64 + kb);
#pragma unroll
      for (int m = 0; m < 2; ++m) {
        v16bf a = *(const v16bf*)(wf + L3F + ((m * 2 + kc) * 32 + lane) * 16);
        acc3[m] = wmma_bf16(a, bb, acc3[m]);
      }
    }
#pragma unroll
    for (int m = 0; m < 2; ++m)
#pragma unroll
      for (int r = 0; r < 8; ++r) {
        int row = m * 16 + r + hi;
        float fb = prm[G3 + row] * RS * prm[B3 + row] + prm[T3 + row];
        Dbuf[row * 128 + w * 16 + cb] = fmaxf(acc3[m][r] + fb, 0.f);
      }
  }
  __syncthreads();
  if (tid < 128) {
    int ch = tid & 31, q = tid >> 5;
    const float* dp = Dbuf + ch * 128 + q * 32;
    float mv = dp[0];
    for (int s = 1; s < 32; ++s) mv = fmaxf(mv, dp[s]);
    sfout[(size_t)(b * 32 + ch) * Q + qbase + q] = mv;
  }
}

// ---------------------------------------------------------------------------
// Matching heads (thread-per-column VALU; weights staged in LDS)
// ---------------------------------------------------------------------------
__global__ __launch_bounds__(256) void matching_kernel(
    const float* __restrict__ prm, const float* __restrict__ sf,
    const float* __restrict__ lf, float* __restrict__ out) {
  __shared__ float sW[2][1024], sb[2][32], sbt[2][32], sg[2][32], sWp[2][64],
      sbp[2][2];
  int tid = threadIdx.x;
  for (int i = tid; i < 1024; i += 256) {
    sW[0][i] = prm[2240 + i];
    sW[1][i] = prm[3426 + i];
  }
  if (tid < 32) {
    sb[0][tid] = prm[3264 + tid]; sbt[0][tid] = prm[3296 + tid];
    sg[0][tid] = prm[3328 + tid];
    sb[1][tid] = prm[4450 + tid]; sbt[1][tid] = prm[4482 + tid];
    sg[1][tid] = prm[4514 + tid];
  }
  if (tid < 64) { sWp[0][tid] = prm[3360 + tid]; sWp[1][tid] = prm[4546 + tid]; }
  if (tid < 2)  { sbp[0][tid] = prm[3424 + tid]; sbp[1][tid] = prm[4610 + tid]; }
  __syncthreads();

  int idx = blockIdx.x * 256 + tid;
  int b = idx / 4608, c = idx - b * 4608;
  const float RS = rsqrtf(1.f + 1e-5f);
  float x[32];
  if (c < 1536)
    for (int ch = 0; ch < 32; ++ch) x[ch] = sf[(size_t)(b * 32 + ch) * 1536 + c];
  else {
    int cl = c - 1536;
    for (int ch = 0; ch < 32; ++ch) x[ch] = lf[(size_t)(b * 32 + ch) * 3072 + cl];
  }
  for (int set = 0; set < 2; ++set) {
    float y[32];
    for (int o = 0; o < 32; ++o) {
      float d = 0.f;
      for (int k = 0; k < 32; ++k) d += sW[set][o * 32 + k] * x[k];
      y[o] = fmaxf((d + sb[set][o]) * (sg[set][o] * RS) + sbt[set][o], 0.f);
    }
    float m0 = sbp[set][0], m1 = sbp[set][1];
    for (int k = 0; k < 32; ++k) {
      m0 += sWp[set][k] * y[k];
      m1 += sWp[set][32 + k] * y[k];
    }
    size_t o = (size_t)(set ? OUT_MSS : OUT_MS) + (size_t)(b * 4608 + c) * 2;
    out[o] = m0;
    out[o + 1] = m1;
  }
}

// ---------------------------------------------------------------------------
// Agg heads + reshape into bf16 combine_feature [b][col(256)][576]
// ---------------------------------------------------------------------------
__global__ __launch_bounds__(256) void agg_cf_kernel(
    const float* __restrict__ prm, const float* __restrict__ sf,
    const float* __restrict__ lf, __bf16* __restrict__ cf) {
  __shared__ float sW[4][1024], sb[4][32], sbt[4][32], sg[4][32];
  int tid = threadIdx.x;
  for (int i = tid; i < 1024; i += 256) {
    sW[0][i] = prm[182437 + i];
    sW[1][i] = prm[183557 + i];
    sW[2][i] = prm[0 + i];
    sW[3][i] = prm[1120 + i];
  }
  if (tid < 32) {
    const int BO[4] = {183461, 184581, 1024, 2144};
    const int TO[4] = {183493, 184613, 1056, 2176};
    const int GO[4] = {183525, 184645, 1088, 2208};
    for (int s = 0; s < 4; ++s) {
      sb[s][tid] = prm[BO[s] + tid];
      sbt[s][tid] = prm[TO[s] + tid];
      sg[s][tid] = prm[GO[s] + tid];
    }
  }
  __syncthreads();

  int idx = blockIdx.x * 256 + tid;
  int b = idx / 4608, c = idx - b * 4608;
  bool surf = c < 1536;
  int s0 = surf ? 0 : 2;
  const float RS = rsqrtf(1.f + 1e-5f);
  float x[32];
  if (surf)
    for (int ch = 0; ch < 32; ++ch) x[ch] = sf[(size_t)(b * 32 + ch) * 1536 + c];
  else {
    int cl = c - 1536;
    for (int ch = 0; ch < 32; ++ch) x[ch] = lf[(size_t)(b * 32 + ch) * 3072 + cl];
  }
  float y[32];
  for (int o = 0; o < 32; ++o) {
    float d = 0.f;
    for (int k = 0; k < 32; ++k) d += sW[s0][o * 32 + k] * x[k];
    y[o] = fmaxf((d + sb[s0][o]) * (sg[s0][o] * RS) + sbt[s0][o], 0.f);
  }
  int j, p, rbase, rstride;
  if (surf) { j = c >> 8; p = c & 255; rbase = j; rstride = 6; }
  else { int cl = c - 1536; j = cl >> 8; p = cl & 255; rbase = 192 + j; rstride = 12; }
  __bf16* dst = cf + (size_t)(b * 256 + p) * 576;
  int s1 = s0 + 1;
  for (int o = 0; o < 32; ++o) {
    float d = 0.f;
    for (int k = 0; k < 32; ++k) d += sW[s1][o * 32 + k] * y[k];
    float z = fmaxf((d + sb[s1][o]) * (sg[s1][o] * RS) + sbt[s1][o], 0.f);
    dst[rbase + o * rstride] = (__bf16)z;
  }
}

// ---------------------------------------------------------------------------
// Proposal head: 576->128 (+af) ->128 ->128 ->97, all WMMA.
// Grid: 8 workgroups (b x half of 256 cols). 32KB LDS ping-pong buffer.
// ---------------------------------------------------------------------------
__global__ __launch_bounds__(256) void proposal_kernel(
    const float* __restrict__ prm, const float* __restrict__ af,
    const __bf16* __restrict__ wf, const __bf16* __restrict__ cf,
    float* __restrict__ out) {
  __shared__ __align__(16) unsigned char smem[32768];
  __bf16* Pbuf = (__bf16*)smem;  // [128 local cols][128 ch]
  const int tid = threadIdx.x, w = tid >> 5, lane = tid & 31;
  const int b = blockIdx.x >> 1;
  const int c0 = (blockIdx.x & 1) * 128;
  const float RS = rsqrtf(1.f + 1e-5f);
  const int cb = lane & 15, kh = (lane >> 4) * 16, hi = (lane >> 4) * 8;

  // P1: K=576 (18 chunks), B-fragments straight from global combine_feature
  v8f acc[8] = {};
  for (int kc = 0; kc < 18; ++kc) {
    v16bf a = *(const v16bf*)(wf + WH_P1 + ((w * 18 + kc) * 32 + lane) * 16);
    int kb = kc * 32 + kh;
#pragma unroll
    for (int nt = 0; nt < 8; ++nt) {
      int col = c0 + nt * 16 + cb;
      v16bf bb = *(const v16bf*)(cf + (size_t)(b * 256 + col) * 576 + kb);
      acc[nt] = wmma_bf16(a, bb, acc[nt]);
    }
  }
  {
    float fb[8];
#pragma unroll
    for (int r = 0; r < 8; ++r) {
      int row = w * 16 + r + hi;
      fb[r] = prm[78596 + row] * RS * prm[78340 + row] + prm[78468 + row];
    }
#pragma unroll
    for (int nt = 0; nt < 8; ++nt)
#pragma unroll
      for (int r = 0; r < 8; ++r) {
        int row = w * 16 + r + hi;
        int col = c0 + nt * 16 + cb;
        float v = fmaxf(acc[nt][r] + fb[r], 0.f) +
                  af[(size_t)(b * 128 + row) * 256 + col];
        Pbuf[(nt * 16 + cb) * 128 + row] = (__bf16)v;
      }
  }
  __syncthreads();

  // P2 and P3: 128x128x128, LDS ping-pong with barriers
  const int WOFF[2] = {WH_P2, WH_P3};
  const int BOFF[2] = {95108, 111876};
  const int TOFF[2] = {95236, 112004};
  const int GOFF[2] = {95364, 112132};
  for (int L = 0; L < 2; ++L) {
    v8f a2[8] = {};
#pragma unroll
    for (int kc = 0; kc < 4; ++kc) {
      v16bf a = *(const v16bf*)(wf + WOFF[L] + ((w * 4 + kc) * 32 + lane) * 16);
      int kb = kc * 32 + kh;
#pragma unroll
      for (int nt = 0; nt < 8; ++nt) {
        v16bf bb = *(const v16bf*)(Pbuf + (nt * 16 + cb) * 128 + kb);
        a2[nt] = wmma_bf16(a, bb, a2[nt]);
      }
    }
    __syncthreads();  // all waves done reading before overwrite
    float fb[8];
#pragma unroll
    for (int r = 0; r < 8; ++r) {
      int row = w * 16 + r + hi;
      fb[r] = prm[GOFF[L] + row] * RS * prm[BOFF[L] + row] + prm[TOFF[L] + row];
    }
#pragma unroll
    for (int nt = 0; nt < 8; ++nt)
#pragma unroll
      for (int r = 0; r < 8; ++r) {
        int row = w * 16 + r + hi;
        float v = fmaxf(a2[nt][r] + fb[r], 0.f);
        Pbuf[(nt * 16 + cb) * 128 + row] = (__bf16)v;
      }
    __syncthreads();
  }

  // Final: 97(x pad 112) x 128, waves 0..6
  if (w < 7) {
    v8f a4[8] = {};
#pragma unroll
    for (int kc = 0; kc < 4; ++kc) {
      v16bf a = *(const v16bf*)(wf + WH_PF + ((w * 4 + kc) * 32 + lane) * 16);
      int kb = kc * 32 + kh;
#pragma unroll
      for (int nt = 0; nt < 8; ++nt) {
        v16bf bb = *(const v16bf*)(Pbuf + (nt * 16 + cb) * 128 + kb);
        a4[nt] = wmma_bf16(a, bb, a4[nt]);
      }
    }
#pragma unroll
    for (int nt = 0; nt < 8; ++nt)
#pragma unroll
      for (int r = 0; r < 8; ++r) {
        int row = w * 16 + r + hi;
        if (row < 97) {
          float v = a4[nt][r] + prm[124676 + row];
          out[(size_t)(b * 97 + row) * 256 + c0 + nt * 16 + cb] = v;
        }
      }
  }
}

// ---------------------------------------------------------------------------
// surface_sem_pred: concat(sem_z, sem_xy) along axis 1
// ---------------------------------------------------------------------------
__global__ void sem_kernel(const float* __restrict__ semz,
                           const float* __restrict__ semxy,
                           float* __restrict__ out) {
  int idx = blockIdx.x * 256 + threadIdx.x;
  int b = idx / 36864;
  int rem = idx - b * 36864;
  float v = (rem < 18432) ? semz[b * 18432 + rem] : semxy[b * 18432 + rem - 18432];
  out[OUT_SEM + idx] = v;
}

// ---------------------------------------------------------------------------
extern "C" void kernel_launch(void* const* d_in, const int* in_sizes, int n_in,
                              void* d_out, int out_size, void* d_ws, size_t ws_size,
                              hipStream_t stream) {
  (void)in_sizes; (void)n_in; (void)out_size; (void)ws_size;
  const float* prm  = (const float*)d_in[0];
  const float* afz  = (const float*)d_in[1];
  const float* afxy = (const float*)d_in[2];
  const float* afl  = (const float*)d_in[3];
  const float* af   = (const float*)d_in[4];
  const float* zc   = (const float*)d_in[5];
  const float* xyc  = (const float*)d_in[6];
  const float* lc   = (const float*)d_in[7];
  const float* semz = (const float*)d_in[8];
  const float* semxy= (const float*)d_in[9];
  const float* prop = (const float*)d_in[10];
  float* out = (float*)d_out;
  char* ws = (char*)d_ws;
  __bf16* wfrag = (__bf16*)ws;
  float* sf = (float*)(ws + WSB_SF);
  float* lf = (float*)(ws + WSB_LF);
  __bf16* cf = (__bf16*)(ws + WSB_CF);

  auto prep = [&](int woff, int goff, int cout, int cin, int MT, int KC, int dh) {
    int tot = MT * KC * 512;
    prep_wfrag<<<(tot + 255) / 256, 256, 0, stream>>>(
        prm + woff, goff >= 0 ? prm + goff : nullptr, cout, cin, MT, KC,
        wfrag + dh);
  };
  prep(153989, 171781, 128, 137, 8, 5, WH_SURF_L1);
  prep(124773, 143333, 128, 143, 8, 5, WH_LINE_L1);
  prep(171909, 180229, 64, 128, 4, 4, WH_SURF_L2);
  prep(143461, 151781, 64, 128, 4, 4, WH_LINE_L2);
  prep(180293, 182405, 32, 64, 2, 2, WH_SURF_L3);
  prep(151845, 153957, 32, 64, 2, 2, WH_LINE_L3);
  prep(4612,   78596,  128, 576, 8, 18, WH_P1);
  prep(78724,  95364,  128, 128, 8, 4, WH_P2);
  prep(95492,  112132, 128, 128, 8, 4, WH_P3);
  prep(112260, -1,     97,  128, 7, 4, WH_PF);

  sa_kernel<0><<<4 * 384, 256, 0, stream>>>(prm, zc, xyc, afz, afxy, prop, wfrag, sf);
  sa_kernel<1><<<4 * 768, 256, 0, stream>>>(prm, lc, nullptr, afl, nullptr, prop, wfrag, lf);
  matching_kernel<<<72, 256, 0, stream>>>(prm, sf, lf, out);
  agg_cf_kernel<<<72, 256, 0, stream>>>(prm, sf, lf, cf);
  proposal_kernel<<<8, 256, 0, stream>>>(prm, af, wfrag, cf, out);
  sem_kernel<<<576, 256, 0, stream>>>(semz, semxy, out);
}
